// PointerAttnMoE_22454089024213
// MI455X (gfx1250) — matrix operand order repaired and checked by reading, not compiled, not run on hardware
//
#include <hip/hip_runtime.h>
#include <hip/hip_bf16.h>
#include <math.h>

// Problem constants (from the reference)
constexpr int B_  = 16;
constexpr int S_  = 256;
constexpr int G_  = 2048;
constexpr int D_  = 512;
constexpr int H_  = 8;
constexpr int DK_ = 64;
constexpr int E_  = 8;

typedef __attribute__((ext_vector_type(16))) __bf16 v16bf;
typedef __attribute__((ext_vector_type(8)))  float  v8f;
typedef __attribute__((ext_vector_type(2)))  __bf16 v2bf;

#if defined(__has_builtin)
#if __has_builtin(__builtin_amdgcn_cvt_pk_bf16_f32)
#define HAVE_PK_BF16 1
#endif
#endif

// Packed f32x2 -> bf16x2 (one v_cvt_pk_bf16_f32 when available)
__device__ __forceinline__ unsigned f2bf_pk(float lo, float hi) {
#ifdef HAVE_PK_BF16
    v2bf p = __builtin_amdgcn_cvt_pk_bf16_f32(lo, hi);
    return __builtin_bit_cast(unsigned, p);
#else
    unsigned short a = __builtin_bit_cast(unsigned short, (__bf16)lo);
    unsigned short b = __builtin_bit_cast(unsigned short, (__bf16)hi);
    return (unsigned)a | ((unsigned)b << 16);
#endif
}
__device__ __forceinline__ unsigned short f2bf(float f) {
    return __builtin_bit_cast(unsigned short, (__bf16)f);
}

// GEMM tiling: block 256x64, K-step 32 (== WMMA K), 8 waves, wave tile 32x64.
// Double-buffered LDS: one barrier per K-step, staging overlaps WMMA.
constexpr int BM = 256, BN = 64, BK = 32, LD = 40;      // LD: padded LDS row (bf16)
constexpr int ASZ = BM * LD, BSZ = BN * LD;

union Frag { v16bf v; uint4 u[2]; };

// Build the 2 A-fragments / 4 B-fragments for a wave and run 8 WMMAs.
__device__ __forceinline__ void wave_mma(const unsigned short* __restrict__ as,
                                         const unsigned short* __restrict__ bs,
                                         int wm, int half, int l16, v8f acc[2][4])
{
    Frag a[2], b[4];
    const int ko = half * 8;
#pragma unroll
    for (int i = 0; i < 2; ++i) {
        const int m = wm + i * 16 + l16;
        a[i].u[0] = *(const uint4*)(&as[m * LD + ko]);
        a[i].u[1] = *(const uint4*)(&as[m * LD + 16 + ko]);
    }
#pragma unroll
    for (int j = 0; j < 4; ++j) {
        const int n = j * 16 + l16;
        b[j].u[0] = *(const uint4*)(&bs[n * LD + half * 8]);
        b[j].u[1] = *(const uint4*)(&bs[n * LD + 16 + half * 8]);
    }
#pragma unroll
    for (int i = 0; i < 2; ++i)
#pragma unroll
        for (int j = 0; j < 4; ++j)
            acc[i][j] = __builtin_amdgcn_wmma_f32_16x16x32_bf16(
                false, a[i].v, false, b[j].v, (short)0, acc[i][j], false, false);
}

// ---------------------------------------------------------------------------
// Generic batched GEMM: C = alpha * A(fp32, MxK) * B(fp32)   with bf16 WMMA.
//   B_NXK=false : B is row-major K x N (ldb = row stride over k)
//   B_NXK=true  : B is row-major N x K (ldb = row stride over n)  (i.e. B^T)
// Batch index bz decomposes as (bz/zdiv, bz%zdiv) with two strides per tensor.
// Preconditions: M%256==0, N%64==0, K%32==0, lda/ldb/ldc%4==0.
// ---------------------------------------------------------------------------
template <bool B_NXK>
__global__ __launch_bounds__(256)
void gemm_bf16(const float* __restrict__ A, const float* __restrict__ Bg,
               float* __restrict__ C,
               int M, int N, int K, int lda, int ldb, int ldc,
               int zdiv,
               long long sA, long long sA2,
               long long sB, long long sB2,
               long long sC, long long sC2,
               float alpha)
{
    __shared__ unsigned short As[2 * ASZ];
    __shared__ unsigned short Bs[2 * BSZ];

    const int t    = threadIdx.x;
    const int lane = t & 31;
    const int wid  = t >> 5;
    const int half = lane >> 4;
    const int l16  = lane & 15;
    const int wm   = wid * 32;          // wave tile rows [wm, wm+32)

    const int bz = blockIdx.z;
    const long long z0 = bz / zdiv, z1 = bz % zdiv;
    A  += z0 * sA + z1 * sA2 + (long long)(blockIdx.y * BM) * lda;
    Bg += z0 * sB + z1 * sB2;
    C  += z0 * sC + z1 * sC2 + (long long)(blockIdx.y * BM) * ldc + blockIdx.x * BN;
    const int n0 = blockIdx.x * BN;

    // tile stagers (global fp32 -> LDS bf16)
    auto stageA = [&](int k0, int buf) {
        unsigned short* as = As + buf * ASZ;
        const int row = t >> 3;          // 0..31
        const int col = (t & 7) * 4;     // 0,4,..,28
        if (k0 + BK < K)
            __builtin_prefetch((const void*)(A + (long long)row * lda + k0 + BK), 0, 1);
#pragma unroll
        for (int i = 0; i < 8; ++i) {
            const int m = row + i * 32;
            const float4 v = *(const float4*)(A + (long long)m * lda + k0 + col);
            unsigned* d = (unsigned*)&as[m * LD + col];
            d[0] = f2bf_pk(v.x, v.y);
            d[1] = f2bf_pk(v.z, v.w);
        }
    };
    auto stageB = [&](int k0, int buf) {
        unsigned short* bs = Bs + buf * BSZ;
        if constexpr (B_NXK) {
            const int n   = t >> 3;          // 0..31
            const int col = (t & 7) * 4;     // k offset
#pragma unroll
            for (int i = 0; i < 2; ++i) {
                const int nn = n + i * 32;
                const float4 v = *(const float4*)(Bg + (long long)(n0 + nn) * ldb + k0 + col);
                unsigned* d = (unsigned*)&bs[nn * LD + col];
                d[0] = f2bf_pk(v.x, v.y);
                d[1] = f2bf_pk(v.z, v.w);
            }
        } else {
            const int k   = t >> 4;          // 0..15
            const int col = (t & 15) * 4;    // n offset
#pragma unroll
            for (int i = 0; i < 2; ++i) {
                const int kk = k + i * 16;
                const float4 v = *(const float4*)(Bg + (long long)(k0 + kk) * ldb + n0 + col);
                bs[(col + 0) * LD + kk] = f2bf(v.x);
                bs[(col + 1) * LD + kk] = f2bf(v.y);
                bs[(col + 2) * LD + kk] = f2bf(v.z);
                bs[(col + 3) * LD + kk] = f2bf(v.w);
            }
        }
    };

    v8f acc[2][4] = {};
    const int nt = K / BK;

    stageA(0, 0);
    stageB(0, 0);
    __syncthreads();
    for (int kt = 0; kt < nt; ++kt) {
        const int cur = kt & 1;
        if (kt + 1 < nt) {               // stage next tile while computing current
            stageA((kt + 1) * BK, cur ^ 1);
            stageB((kt + 1) * BK, cur ^ 1);
        }
        wave_mma(As + cur * ASZ, Bs + cur * BSZ, wm, half, l16, acc);
        __syncthreads();
    }

    // store C (16x16 f32 C/D layout: VGPR r -> M = r + 8*half)
#pragma unroll
    for (int i = 0; i < 2; ++i)
#pragma unroll
        for (int j = 0; j < 4; ++j) {
            const int mBase = wm + i * 16 + half * 8;
            const int nCol  = j * 16 + l16;
#pragma unroll
            for (int r = 0; r < 8; ++r)
                C[(long long)(mBase + r) * ldc + nCol] = alpha * acc[i][j][r];
        }
}

// ---------------------------------------------------------------------------
// MoE GEMM: C[m,:] = sum_e gate[m,e] * (x[m,:] @ W_e)
// One K = E*D GEMM with the (tile-uniform) expert gate folded into the A tile.
// Wb is expert_w flattened: (E*D) x D row-major.
// ---------------------------------------------------------------------------
__global__ __launch_bounds__(256)
void moe_gemm(const float* __restrict__ X, const float* __restrict__ Gt,
              const float* __restrict__ Wb, float* __restrict__ C)
{
    __shared__ unsigned short As[2 * ASZ];
    __shared__ unsigned short Bs[2 * BSZ];

    const int t    = threadIdx.x;
    const int lane = t & 31, wid = t >> 5;
    const int half = lane >> 4, l16 = lane & 15;
    const int wm = wid * 32;
    const int m0 = blockIdx.y * BM, n0 = blockIdx.x * BN;

    auto stageA = [&](int k0, int buf) {
        unsigned short* as = As + buf * ASZ;
        const int e  = k0 >> 9;          // expert index (tile-uniform: BK | D_)
        const int d0 = k0 & (D_ - 1);
        const int row = t >> 3;
        const int col = (t & 7) * 4;
#pragma unroll
        for (int i = 0; i < 8; ++i) {
            const int mr = row + i * 32;
            const int m  = m0 + mr;
            const float g  = Gt[m * E_ + e];
            const float4 v = *(const float4*)(X + (long long)m * D_ + d0 + col);
            unsigned* d = (unsigned*)&as[mr * LD + col];
            d[0] = f2bf_pk(v.x * g, v.y * g);
            d[1] = f2bf_pk(v.z * g, v.w * g);
        }
    };
    auto stageB = [&](int k0, int buf) {
        unsigned short* bs = Bs + buf * BSZ;
        const int k   = t >> 4;
        const int col = (t & 15) * 4;
#pragma unroll
        for (int i = 0; i < 2; ++i) {
            const int kk = k + i * 16;
            const float4 v = *(const float4*)(Wb + (long long)(k0 + kk) * D_ + n0 + col);
            bs[(col + 0) * LD + kk] = f2bf(v.x);
            bs[(col + 1) * LD + kk] = f2bf(v.y);
            bs[(col + 2) * LD + kk] = f2bf(v.z);
            bs[(col + 3) * LD + kk] = f2bf(v.w);
        }
    };

    v8f acc[2][4] = {};
    const int nt = (E_ * D_) / BK;

    stageA(0, 0);
    stageB(0, 0);
    __syncthreads();
    for (int kt = 0; kt < nt; ++kt) {
        const int cur = kt & 1;
        if (kt + 1 < nt) {
            stageA((kt + 1) * BK, cur ^ 1);
            stageB((kt + 1) * BK, cur ^ 1);
        }
        wave_mma(As + cur * ASZ, Bs + cur * BSZ, wm, half, l16, acc);
        __syncthreads();
    }

#pragma unroll
    for (int i = 0; i < 2; ++i)
#pragma unroll
        for (int j = 0; j < 4; ++j) {
            const int mBase = m0 + wm + i * 16 + half * 8;
            const int nCol  = n0 + j * 16 + l16;
#pragma unroll
            for (int r = 0; r < 8; ++r)
                C[(long long)(mBase + r) * D_ + nCol] = acc[i][j][r];
        }
}

// ---------------------------------------------------------------------------
// Masked softmax over G for rows of scores (B*H*S rows). One wave per row.
// ---------------------------------------------------------------------------
__global__ __launch_bounds__(256)
void softmax_mask(float* __restrict__ scores, const unsigned char* __restrict__ mask)
{
    const int lane = threadIdx.x & 31;
    const int row  = blockIdx.x * 8 + (threadIdx.x >> 5);
    const int b    = row / (H_ * S_);
    float* p = scores + (long long)row * G_;
    const unsigned char* mrow = mask + (long long)b * G_;

    float vals[G_ / 32];
    float mx = -3.4e38f;
#pragma unroll
    for (int i = 0; i < G_ / 32; ++i) {
        const int g = lane + i * 32;
        float v = p[g];
        if (mrow[g]) v = -1.0e9f;
        vals[i] = v;
        mx = fmaxf(mx, v);
    }
#pragma unroll
    for (int o = 16; o > 0; o >>= 1) mx = fmaxf(mx, __shfl_xor(mx, o, 32));
    float sum = 0.f;
#pragma unroll
    for (int i = 0; i < G_ / 32; ++i) { const float e = __expf(vals[i] - mx); vals[i] = e; sum += e; }
#pragma unroll
    for (int o = 16; o > 0; o >>= 1) sum += __shfl_xor(sum, o, 32);
    const float inv = 1.0f / sum;
#pragma unroll
    for (int i = 0; i < G_ / 32; ++i) p[lane + i * 32] = vals[i] * inv;
}

// ---------------------------------------------------------------------------
// Top-2 gating: one thread per row.
// ---------------------------------------------------------------------------
__global__ void gate_kernel(const float* __restrict__ X, const float* __restrict__ Wg,
                            float* __restrict__ Gt)
{
    const int n = blockIdx.x * blockDim.x + threadIdx.x;
    if (n >= B_ * S_) return;
    const float* x = X + (long long)n * D_;
    float l[E_] = {};
    for (int d = 0; d < D_; ++d) {
        const float xv = x[d];
#pragma unroll
        for (int e = 0; e < E_; ++e) l[e] += xv * Wg[d * E_ + e];
    }
    int i0 = 0;
#pragma unroll
    for (int e = 1; e < E_; ++e) if (l[e] > l[i0]) i0 = e;
    int i1 = (i0 == 0) ? 1 : 0;
#pragma unroll
    for (int e = 0; e < E_; ++e) if (e != i0 && l[e] > l[i1]) i1 = e;
    const float g0 = 1.0f / (1.0f + __expf(l[i1] - l[i0]));
    const float g1 = 1.0f - g0;
#pragma unroll
    for (int e = 0; e < E_; ++e) Gt[n * E_ + e] = 0.f;
    Gt[n * E_ + i0] = g0;
    Gt[n * E_ + i1] = g1;
}

// ---------------------------------------------------------------------------
// Repack Wq/Wk/Wv: (H, D, DK) -> (D, H*DK) row-major (K x N for the GEMM).
// ---------------------------------------------------------------------------
__global__ void pack_qkv(const float* __restrict__ Wq, const float* __restrict__ Wk,
                         const float* __restrict__ Wv, float* __restrict__ out)
{
    const int idx = blockIdx.x * 256 + threadIdx.x;
    if (idx >= 3 * D_ * D_) return;
    const int which = idx / (D_ * D_);
    const int r  = idx % (D_ * D_);
    const int d  = r / D_;        // GEMM k
    const int n  = r % D_;        // GEMM n = h*DK + kk
    const int h  = n / DK_, kk = n % DK_;
    const float* W = (which == 0) ? Wq : (which == 1) ? Wk : Wv;
    out[idx] = W[((long long)h * D_ + d) * DK_ + kk];
}

// ---------------------------------------------------------------------------
extern "C" void kernel_launch(void* const* d_in, const int* in_sizes, int n_in,
                              void* d_out, int out_size, void* d_ws, size_t ws_size,
                              hipStream_t stream)
{
    const float* query     = (const float*)d_in[0];   // (B,S,D)
    const float* key       = (const float*)d_in[1];   // (B,G,D)
    const float* value     = (const float*)d_in[2];   // (B,G,D)
    const float* logit_key = (const float*)d_in[3];   // (B,G,D)
    const unsigned char* attn_mask = (const unsigned char*)d_in[4]; // (B,G) bool
    // d_in[5..7] = Wq,Wk,Wv (H,D,DK); d_in[8] = Wo (H,DK,D) == (D,D) K x N already
    const float* Wo        = (const float*)d_in[8];
    const float* w_gate    = (const float*)d_in[9];   // (D,E)
    const float* expert_w  = (const float*)d_in[10];  // (E,D,D) == (E*D, D) K x N
    float* out = (float*)d_out;
    float* ws  = (float*)d_ws;

    // Workspace layout (fp32 elements)
    const long long o_wq = 0;                                   // 512*512
    const long long o_wk = o_wq + (long long)D_ * D_;
    const long long o_wv = o_wk + (long long)D_ * D_;
    const long long o_qp = o_wv + (long long)D_ * D_;           // B*S*D
    const long long o_kp = o_qp + (long long)B_ * S_ * D_;      // B*G*D
    const long long o_vp = o_kp + (long long)B_ * G_ * D_;      // B*G*D
    const long long o_sc = o_vp + (long long)B_ * G_ * D_;      // B*H*S*G
    const long long o_hd = o_sc + (long long)B_ * H_ * S_ * G_; // B*S*D
    const long long o_gl = o_hd + (long long)B_ * S_ * D_;      // B*S*D
    const long long o_gt = o_gl + (long long)B_ * S_ * D_;      // (B*S)*E
    const long long o_gm = o_qp;                                // alias: Qp dead after scores

    const long long sSD = (long long)S_ * D_;   // 131072
    const long long sGD = (long long)G_ * D_;   // 1048576
    const long long sSG = (long long)S_ * G_;   // 524288

    const float inv_sqrt_dk = 0.125f;                 // 1/sqrt(64)
    const float inv_sqrt_d  = 0.044194173824159216f;  // 1/sqrt(512)

    // 1) repack projection weights to (D, D) K x N
    pack_qkv<<<(3 * D_ * D_ + 255) / 256, 256, 0, stream>>>(
        (const float*)d_in[5], (const float*)d_in[6], (const float*)d_in[7], ws + o_wq);

    // 2) Q = query @ Wq_all   (per batch: 256x512x512)
    gemm_bf16<false><<<dim3(D_ / BN, S_ / BM, B_), 256, 0, stream>>>(
        query, ws + o_wq, ws + o_qp, S_, D_, D_, D_, D_, D_,
        1, sSD, 0, 0, 0, sSD, 0, 1.0f);

    // 3) K = key @ Wk_all     (per batch: 2048x512x512)
    gemm_bf16<false><<<dim3(D_ / BN, G_ / BM, B_), 256, 0, stream>>>(
        key, ws + o_wk, ws + o_kp, G_, D_, D_, D_, D_, D_,
        1, sGD, 0, 0, 0, sGD, 0, 1.0f);

    // 4) V = value @ Wv_all
    gemm_bf16<false><<<dim3(D_ / BN, G_ / BM, B_), 256, 0, stream>>>(
        value, ws + o_wv, ws + o_vp, G_, D_, D_, D_, D_, D_,
        1, sGD, 0, 0, 0, sGD, 0, 1.0f);

    // 5) scores = Q @ K^T / sqrt(DK)   batches (b,h): 256x2048x64, B is N x K
    gemm_bf16<true><<<dim3(G_ / BN, S_ / BM, B_ * H_), 256, 0, stream>>>(
        ws + o_qp, ws + o_kp, ws + o_sc, S_, G_, DK_, D_, D_, G_,
        H_, sSD, DK_, sGD, DK_, (long long)H_ * sSG, sSG, inv_sqrt_dk);

    // 6) masked softmax over G (in place)
    softmax_mask<<<(B_ * H_ * S_) / 8, 256, 0, stream>>>(ws + o_sc, attn_mask);

    // 7) heads = attn @ V   batches (b,h): 256x64x2048, B is K x N
    gemm_bf16<false><<<dim3(DK_ / BN, S_ / BM, B_ * H_), 256, 0, stream>>>(
        ws + o_sc, ws + o_vp, ws + o_hd, S_, DK_, G_, G_, D_, D_,
        H_, (long long)H_ * sSG, sSG, sGD, DK_, sSD, DK_, 1.0f);

    // 8) glimpse = heads @ Wo   (per batch: 256x512x512; Wo already K x N)
    gemm_bf16<false><<<dim3(D_ / BN, S_ / BM, B_), 256, 0, stream>>>(
        ws + o_hd, Wo, ws + o_gl, S_, D_, D_, D_, D_, D_,
        1, sSD, 0, 0, 0, sSD, 0, 1.0f);

    // 9) top-2 gates
    gate_kernel<<<(B_ * S_ + 255) / 256, 256, 0, stream>>>(ws + o_gl, w_gate, ws + o_gt);

    // 10) gm = sum_e gate_e * (glimpse @ W_e)   (4096 x 512, K = E*D = 4096)
    moe_gemm<<<dim3(D_ / BN, (B_ * S_) / BM, 1), 256, 0, stream>>>(
        ws + o_gl, ws + o_gt, expert_w, ws + o_gm);

    // 11) out = gm @ logit_key^T / sqrt(D)   (per batch: 256x2048x512, B is N x K)
    gemm_bf16<true><<<dim3(G_ / BN, S_ / BM, B_), 256, 0, stream>>>(
        ws + o_gm, logit_key, out, S_, G_, D_, D_, D_, G_,
        1, sSD, 0, sGD, 0, sSG, 0, inv_sqrt_d);
}
